// WindowTransformerBlock_45741401702773
// MI455X (gfx1250) — matrix operand order repaired and checked by reading, not compiled, not run on hardware
//
#include <hip/hip_runtime.h>
#include <hip/hip_bf16.h>
#include <math.h>
#include <stdint.h>

typedef __attribute__((ext_vector_type(16))) _Float16 v16h;
typedef __attribute__((ext_vector_type(8)))  float    v8f;

union AFrag { uint4 u[2]; v16h h; };
union F8    { v8f  v; float f[8]; };

#define WMMA_F16(a, b, c) \
  __builtin_amdgcn_wmma_f32_16x16x32_f16(false, (a), false, (b), (short)0, (c), false, false)

// ---------------------------------------------------------------------------
// workspace layout (bytes)
// ---------------------------------------------------------------------------
static constexpr size_t QKV_E  = 50331648;   // elements in each of q / k / vT (2048*12*64*32)
static constexpr size_t HW_OFF = 301989888;  // hwin (f16), later reused as attn_out
static constexpr size_t M1_OFF = 100663296;  // MLP hidden chunk (reuses k+vT region)
static constexpr size_t W_OFF  = 402653184;  // f16 transposed weights
static constexpr size_t WQKV_E  = 0;              // [1152][384]
static constexpr size_t WPROJ_E = 442368;         // [384][384]
static constexpr size_t WFC1_E  = 442368 + 147456;// [1536][384]
static constexpr size_t WFC2_E  = WFC1_E + 589824;// [384][1536]
static constexpr size_t BT_OFF  = W_OFF + 3538944;// f32 bias table [12][64][64]

// ---------------------------------------------------------------------------
// fragment loaders
// ---------------------------------------------------------------------------
__device__ __forceinline__ v16h load_fragA(const _Float16* base, int row0, int pitch) {
  const int l  = threadIdx.x & 31;
  const int m  = row0 + (l & 15);
  const int bk = (l >> 4) * 8;          // K base 0 / 8 per lane half
  AFrag fr;
  fr.u[0] = *(const uint4*)(base + (size_t)m * pitch + bk);        // K = bk..bk+7
  fr.u[1] = *(const uint4*)(base + (size_t)m * pitch + bk + 16);   // K = bk+16..bk+23
  return fr.h;
}

// base points at B^T storage: [n][k] with given pitch (in halves)
__device__ __forceinline__ v16h load_fragB(const _Float16* base, int col0, int pitch) {
  const int l  = threadIdx.x & 31;
  const int n  = col0 + (l & 15);
  const int k0 = (l >> 4) * 8;          // K base 0 / 8 per lane half
  AFrag fr;
  fr.u[0] = *(const uint4*)(base + (size_t)n * pitch + k0);
  fr.u[1] = *(const uint4*)(base + (size_t)n * pitch + k0 + 16);
  return fr.h;
}

__device__ __forceinline__ float wave_sum(float v) {
  v += __shfl_xor(v, 16, 32); v += __shfl_xor(v, 8, 32);
  v += __shfl_xor(v, 4, 32);  v += __shfl_xor(v, 2, 32);
  v += __shfl_xor(v, 1, 32);
  return v;
}

__device__ __forceinline__ unsigned lds_addr_u32(const void* p) {
  // generic LDS pointer: [63:32] = shared aperture, [31:0] = wave-relative LDS addr
  return (unsigned)(uintptr_t)p;
}

// ---------------------------------------------------------------------------
// weight convert + transpose: src f32 [K][N] -> dst f16 [N][K]
// ---------------------------------------------------------------------------
__global__ void cvt_tr(const float* __restrict__ src, _Float16* __restrict__ dst,
                       int K, int N) {
  int i = blockIdx.x * 256 + threadIdx.x;
  if (i < K * N) {
    int k = i / N, n = i - k * N;
    dst[(size_t)n * K + k] = (_Float16)src[i];
  }
}

// relative-position bias table: bt[h][m][n], m,n in 8x8 window
__global__ void build_bias(const float* __restrict__ rpb, float* __restrict__ bt) {
  int i = blockIdx.x * 256 + threadIdx.x;       // 12*64*64 = 49152
  if (i < 49152) {
    int h = i >> 12, m = (i >> 6) & 63, n = i & 63;
    int rel = ((m >> 3) - (n >> 3) + 7) * 15 + ((m & 7) - (n & 7) + 7);
    bt[i] = rpb[rel * 12 + h];
  }
}

// ---------------------------------------------------------------------------
// LayerNorm (C = 384), optionally writing in window-partitioned row order
// ---------------------------------------------------------------------------
template <bool WINDOW>
__global__ __launch_bounds__(256) void ln_kernel(const float* __restrict__ src,
                                                 const float* __restrict__ g,
                                                 const float* __restrict__ b,
                                                 _Float16* __restrict__ dst) {
  const int lane = threadIdx.x & 31;
  const int wave = threadIdx.x >> 5;
  const long token = (long)blockIdx.x * 8 + wave;   // 131072 tokens
  const float* row = src + token * 384;

  float v[12];
  float s = 0.f;
#pragma unroll
  for (int j = 0; j < 12; ++j) { v[j] = row[lane + 32 * j]; s += v[j]; }
  s = wave_sum(s);
  const float mu = s * (1.0f / 384.0f);
  float s2 = 0.f;
#pragma unroll
  for (int j = 0; j < 12; ++j) { float d = v[j] - mu; s2 += d * d; }
  s2 = wave_sum(s2);
  const float rs = rsqrtf(s2 * (1.0f / 384.0f) + 1e-5f);

  long drow;
  if (WINDOW) {
    const long bb = token >> 12;
    const int  ll = (int)(token & 4095);
    const int  yy = ll >> 6, xx = ll & 63;
    const int  win = ((yy >> 3) << 3) + (xx >> 3);
    const int  tok = ((yy & 7) << 3) + (xx & 7);
    drow = ((bb << 6) + win) * 64 + tok;
  } else {
    drow = token;
  }
  _Float16* d = dst + drow * 384;
#pragma unroll
  for (int j = 0; j < 12; ++j) {
    const int c = lane + 32 * j;
    d[c] = (_Float16)((v[j] - mu) * rs * g[c] + b[c]);
  }
}

// ---------------------------------------------------------------------------
// Tiled WMMA GEMM: C[M,N] = A[M,K] (f16) * B^T[N,K] (f16), fused epilogues.
// 256 threads / 8 waves; block tile 128x64; wave tile 32x32 (2x2 WMMA).
// Tiles stream into double-buffered LDS with global_load_async_to_lds_b128
// (ASYNCcnt), overlapping the next tile's copy with the current WMMAs.
// MODE 0: QKV  (bias, scale q, split q/k/vT heads; vT transposed)
// MODE 1: PROJ (bias + window-reverse + residual -> f32 d_out)
// MODE 2: FC1  (bias + exact GELU -> f16)
// MODE 3: FC2  (optional bias, accumulate into f32 d_out)
// ---------------------------------------------------------------------------
#define GP 40   // LDS pitch in halves (80B: keeps 16B alignment, spreads banks)

template <int MODE>
__global__ __launch_bounds__(256) void gemm_wmma(const _Float16* __restrict__ A, int lda,
                                                 const _Float16* __restrict__ BT, int ldb,
                                                 int K,
                                                 const float* __restrict__ bias,
                                                 _Float16* __restrict__ outH,
                                                 float* __restrict__ outF,
                                                 const float* __restrict__ resid) {
  __shared__ _Float16 sA[2][128 * GP];
  __shared__ _Float16 sB[2][64 * GP];

  const int tid  = threadIdx.x;
  const int lane = tid & 31;
  const int wave = tid >> 5;
  const int wm   = wave >> 1;           // 0..3
  const int wn   = wave & 1;            // 0..1
  const long m0  = (long)blockIdx.x * 128;
  const int  n0  = blockIdx.y * 64;

  const void* Abase = (const void*)(A + (size_t)m0 * lda);
  const void* Bbase = (const void*)(BT + (size_t)n0 * ldb);

  // per-thread async-copy assignments
  const int ar = tid >> 1, akh = (tid & 1) * 16;  // A: row 0..127, k-half {0,16}
  const int bn = tid >> 2, bkq = (tid & 3) * 8;   // B: n-row 0..63, k-quarter {0,8,16,24}

  auto issue = [&](int kb, int buf) {
    {  // A tile: 32B per thread = two b128 (offset: applies to LDS and global)
      unsigned la = lds_addr_u32(&sA[buf][ar * GP + akh]);
      unsigned va = (unsigned)((ar * lda + kb + akh) * 2);
      asm volatile("global_load_async_to_lds_b128 %0, %1, %2"
                   :: "v"(la), "v"(va), "s"(Abase) : "memory");
      asm volatile("global_load_async_to_lds_b128 %0, %1, %2 offset:16"
                   :: "v"(la), "v"(va), "s"(Abase) : "memory");
    }
    {  // B tile: 16B per thread = one b128
      unsigned lb = lds_addr_u32(&sB[buf][bn * GP + bkq]);
      unsigned vb = (unsigned)((bn * ldb + kb + bkq) * 2);
      asm volatile("global_load_async_to_lds_b128 %0, %1, %2"
                   :: "v"(lb), "v"(vb), "s"(Bbase) : "memory");
    }
  };

  F8 acc[2][2];
#pragma unroll
  for (int a = 0; a < 2; ++a)
#pragma unroll
    for (int b = 0; b < 2; ++b) acc[a][b].v = (v8f)0.0f;

  issue(0, 0);
  int cur = 0;
  for (int kb = 0; kb < K; kb += 32) {
    asm volatile("s_wait_asynccnt 0" ::: "memory");
    __syncthreads();
    if (kb + 32 < K) issue(kb + 32, cur ^ 1);

    v16h af[2], bf[2];
#pragma unroll
    for (int a = 0; a < 2; ++a) af[a] = load_fragA(&sA[cur][0], wm * 32 + a * 16, GP);
#pragma unroll
    for (int b = 0; b < 2; ++b) bf[b] = load_fragB(&sB[cur][0], wn * 32 + b * 16, GP);
#pragma unroll
    for (int a = 0; a < 2; ++a)
#pragma unroll
      for (int b = 0; b < 2; ++b) acc[a][b].v = WMMA_F16(af[a], bf[b], acc[a][b].v);
    cur ^= 1;
  }

  // ---- epilogue ----
  const int hi = lane >> 4, nl = lane & 15;
#pragma unroll
  for (int a = 0; a < 2; ++a)
#pragma unroll
    for (int b = 0; b < 2; ++b)
#pragma unroll
      for (int i = 0; i < 8; ++i) {
        const long m = m0 + wm * 32 + a * 16 + i + 8 * hi;
        const int  n = n0 + wn * 32 + b * 16 + nl;
        float v = acc[a][b].f[i];

        if (MODE == 0) {  // QKV
          v += bias[n];
          const int t = n / 384, rr = n - t * 384;
          const int h = rr >> 5, d = rr & 31;
          const long win = m >> 6; const int tok = (int)(m & 63);
          const size_t hb = (size_t)win * 12 + h;
          if (t == 0)       outH[hb * 2048 + (size_t)tok * 32 + d] =
                                (_Float16)(v * 0.17677669529663689f);
          else if (t == 1)  outH[QKV_E + hb * 2048 + (size_t)tok * 32 + d] = (_Float16)v;
          else              outH[2 * QKV_E + hb * 2048 + (size_t)d * 64 + tok] = (_Float16)v;
        } else if (MODE == 1) {  // PROJ + window reverse + residual
          v += bias[n];
          const long win = m >> 6; const int tok = (int)(m & 63);
          const long bb = win >> 6; const int widx = (int)(win & 63);
          const int wy = widx >> 3, wx = widx & 7, iy = tok >> 3, ix = tok & 7;
          const int l = ((wy * 8 + iy) << 6) + (wx * 8 + ix);
          const size_t off = ((size_t)(bb << 12) + l) * 384 + n;
          outF[off] = resid[off] + v;
        } else if (MODE == 2) {  // FC1 + GELU(exact)
          v += bias[n];
          const float gl = 0.5f * v * (1.0f + erff(v * 0.70710678118654752f));
          outH[(size_t)m * 768 + n] = (_Float16)gl;
        } else {  // FC2 accumulate
          const size_t off = (size_t)m * 384 + n;
          float r = outF[off] + v;
          if (bias) r += bias[n];
          outF[off] = r;
        }
      }
}

// ---------------------------------------------------------------------------
// Window attention: one wave per (window, head). q,k row-major [64,32]; v
// pre-transposed [32,64]. 16 WMMAs for QK^T, bias from precomputed table,
// softmax, P via LDS, 16 WMMAs for P*V.
// grid = (12 heads, 2048 windows)
// ---------------------------------------------------------------------------
__global__ __launch_bounds__(32) void attn_kernel(const _Float16* __restrict__ q,
                                                  const _Float16* __restrict__ k,
                                                  const _Float16* __restrict__ vT,
                                                  const float* __restrict__ btab,
                                                  _Float16* __restrict__ out) {
  __shared__ _Float16 sP[64 * 72];
  const int l = threadIdx.x;
  const int h = blockIdx.x;                     // 0..11
  const size_t win = blockIdx.y;                // 0..2047
  const size_t wh = win * 12 + h;
  const _Float16* qb = q  + wh * 2048;
  const _Float16* kb = k  + wh * 2048;
  const _Float16* vb = vT + wh * 2048;
  const float*    bt = btab + (size_t)h * 4096;

  F8 s[4][4];
#pragma unroll
  for (int r = 0; r < 4; ++r)
#pragma unroll
    for (int c = 0; c < 4; ++c) s[r][c].v = (v8f)0.0f;

  v16h qa[4];
#pragma unroll
  for (int r = 0; r < 4; ++r) qa[r] = load_fragA(qb, r * 16, 32);
#pragma unroll
  for (int c = 0; c < 4; ++c) {
    v16h kf = load_fragB(kb, c * 16, 32);       // k row-major == B^T layout
#pragma unroll
    for (int r = 0; r < 4; ++r) s[r][c].v = WMMA_F16(qa[r], kf, s[r][c].v);
  }

  // bias + row softmax (row of C spans 16 lanes of one vgpr: xor-reduce 8,4,2,1)
  const int hi = l >> 4, nl = l & 15;
#pragma unroll
  for (int r = 0; r < 4; ++r) {
#pragma unroll
    for (int i = 0; i < 8; ++i) {
      const int m = r * 16 + i + 8 * hi;
      const float* brow = bt + m * 64 + nl;
      float mx = -1e30f;
#pragma unroll
      for (int c = 0; c < 4; ++c) {
        s[r][c].f[i] += brow[c * 16];
        mx = fmaxf(mx, s[r][c].f[i]);
      }
      mx = fmaxf(mx, __shfl_xor(mx, 8, 32));
      mx = fmaxf(mx, __shfl_xor(mx, 4, 32));
      mx = fmaxf(mx, __shfl_xor(mx, 2, 32));
      mx = fmaxf(mx, __shfl_xor(mx, 1, 32));
      float sum = 0.f;
#pragma unroll
      for (int c = 0; c < 4; ++c) {
        const float e = __expf(s[r][c].f[i] - mx);
        s[r][c].f[i] = e;
        sum += e;
      }
      sum += __shfl_xor(sum, 8, 32);
      sum += __shfl_xor(sum, 4, 32);
      sum += __shfl_xor(sum, 2, 32);
      sum += __shfl_xor(sum, 1, 32);
      const float inv = 1.0f / sum;
#pragma unroll
      for (int c = 0; c < 4; ++c)
        sP[m * 72 + c * 16 + nl] = (_Float16)(s[r][c].f[i] * inv);
    }
  }
  __syncthreads();

  F8 o[4][2];
#pragma unroll
  for (int r = 0; r < 4; ++r)
#pragma unroll
    for (int b = 0; b < 2; ++b) o[r][b].v = (v8f)0.0f;

#pragma unroll
  for (int kk = 0; kk < 2; ++kk) {
    v16h pa[4];
#pragma unroll
    for (int r = 0; r < 4; ++r) pa[r] = load_fragA(sP + kk * 32, r * 16, 72);
#pragma unroll
    for (int b = 0; b < 2; ++b) {
      v16h vf = load_fragB(vb + kk * 32, b * 16, 64);  // vT == B^T layout
#pragma unroll
      for (int r = 0; r < 4; ++r) o[r][b].v = WMMA_F16(pa[r], vf, o[r][b].v);
    }
  }

#pragma unroll
  for (int r = 0; r < 4; ++r)
#pragma unroll
    for (int b = 0; b < 2; ++b)
#pragma unroll
      for (int i = 0; i < 8; ++i) {
        const int m = r * 16 + i + 8 * hi;
        const int n = b * 16 + nl;
        out[(win * 64 + m) * 384 + h * 32 + n] = (_Float16)o[r][b].f[i];
      }
}

// ---------------------------------------------------------------------------
extern "C" void kernel_launch(void* const* d_in, const int* in_sizes, int n_in,
                              void* d_out, int out_size, void* d_ws, size_t ws_size,
                              hipStream_t stream) {
  const float* x      = (const float*)d_in[0];
  const float* n1g    = (const float*)d_in[1];
  const float* n1b    = (const float*)d_in[2];
  const float* qkv_w  = (const float*)d_in[3];
  const float* qkv_b  = (const float*)d_in[4];
  const float* proj_w = (const float*)d_in[5];
  const float* proj_b = (const float*)d_in[6];
  const float* rpb    = (const float*)d_in[7];
  const float* n2g    = (const float*)d_in[8];
  const float* n2b    = (const float*)d_in[9];
  const float* fc1_w  = (const float*)d_in[10];
  const float* fc1_b  = (const float*)d_in[11];
  const float* fc2_w  = (const float*)d_in[12];
  const float* fc2_b  = (const float*)d_in[13];
  float* out = (float*)d_out;

  char* ws = (char*)d_ws;
  _Float16* qkvT  = (_Float16*)ws;                 // q | k | vT (QKV_E each)
  _Float16* hwin  = (_Float16*)(ws + HW_OFF);      // LN1 (windowed); reused as attn_out
  _Float16* h2    = (_Float16*)ws;                 // reuse q region after attention
  _Float16* m1    = (_Float16*)(ws + M1_OFF);      // MLP hidden chunk (reuse k+vT)
  _Float16* wts   = (_Float16*)(ws + W_OFF);
  _Float16* wqkvT = wts + WQKV_E;                  // [1152][384]
  _Float16* wprojT= wts + WPROJ_E;                 // [384][384]
  _Float16* wfc1T = wts + WFC1_E;                  // [1536][384]
  _Float16* wfc2T = wts + WFC2_E;                  // [384][1536]
  float*    btab  = (float*)(ws + BT_OFF);         // [12][64][64]

  // 1) weights -> f16, transposed to [N][K]; bias table
  cvt_tr<<<(442368 + 255) / 256, 256, 0, stream>>>(qkv_w,  wqkvT,  384, 1152);
  cvt_tr<<<(147456 + 255) / 256, 256, 0, stream>>>(proj_w, wprojT, 384, 384);
  cvt_tr<<<(589824 + 255) / 256, 256, 0, stream>>>(fc1_w,  wfc1T,  384, 1536);
  cvt_tr<<<(589824 + 255) / 256, 256, 0, stream>>>(fc2_w,  wfc2T,  1536, 384);
  build_bias<<<(49152 + 255) / 256, 256, 0, stream>>>(rpb, btab);

  // 2) LN1 + window partition
  ln_kernel<true><<<16384, 256, 0, stream>>>(x, n1g, n1b, hwin);

  // 3) QKV GEMM: [131072,384] x [384,1152]
  gemm_wmma<0><<<dim3(1024, 18), 256, 0, stream>>>(hwin, 384, wqkvT, 384, 384,
                                                   qkv_b, qkvT, nullptr, nullptr);

  // 4) window attention (12 heads x 2048 windows); writes attn_out into hwin slot
  attn_kernel<<<dim3(12, 2048), 32, 0, stream>>>(qkvT, qkvT + QKV_E, qkvT + 2 * QKV_E,
                                                 btab, hwin);

  // 5) proj GEMM + window reverse + residual -> d_out (= x1)
  gemm_wmma<1><<<dim3(1024, 6), 256, 0, stream>>>(hwin, 384, wprojT, 384, 384,
                                                  proj_b, nullptr, out, x);

  // 6) LN2 on x1
  ln_kernel<false><<<16384, 256, 0, stream>>>(out, n2g, n2b, h2);

  // 7) MLP in two 768-wide hidden chunks: d_out += gelu(h2@W1+b1)@W2 + b2
  for (int chunk = 0; chunk < 2; ++chunk) {
    gemm_wmma<2><<<dim3(1024, 12), 256, 0, stream>>>(h2, 384,
                                                     wfc1T + (size_t)chunk * 768 * 384, 384, 384,
                                                     fc1_b + chunk * 768, m1, nullptr, nullptr);
    gemm_wmma<3><<<dim3(1024, 6), 256, 0, stream>>>(m1, 768,
                                                    wfc2T + chunk * 768, 1536, 768,
                                                    chunk == 0 ? fc2_b : nullptr,
                                                    nullptr, out, nullptr);
  }
}
